// MatchLSTM_12627203850958
// MI455X (gfx1250) — matrix operand order repaired and checked by reading, not compiled, not run on hardware
//
#include <hip/hip_runtime.h>
#include <hip/hip_bf16.h>
#include <stdint.h>

// ---------------- problem dims ----------------
#define B_    64
#define SP_   128
#define ST_   128
#define E_    300
#define EP_   320      // E padded to multiple of 32 for WMMA K
#define H_    512
#define H2_   1024     // 2H
#define G4_   2048     // 4H
#define C_    3

typedef __attribute__((ext_vector_type(16))) __bf16 v16bf;
typedef __attribute__((ext_vector_type(8)))  float  v8f;
typedef __attribute__((ext_vector_type(4)))  int    v4i;
typedef unsigned short u16;

struct Chunk16 { uint32_t w[4]; };          // 16 bytes = 8 bf16
union  ABTile  { v16bf v; Chunk16 c[2]; };

#define GLOBAL_AS __attribute__((address_space(1)))
#define LDS_AS    __attribute__((address_space(3)))

#if defined(__has_builtin)
#  if __has_builtin(__builtin_amdgcn_global_load_async_to_lds_b128)
#    define HAVE_ASYNC_LDS 1
#  else
#    define HAVE_ASYNC_LDS 0
#  endif
#else
#  define HAVE_ASYNC_LDS 0
#endif

// stage one 16-byte chunk global -> LDS (async on gfx1250 if builtin available)
__device__ __forceinline__ void stage16(const u16* g, u16* l) {
#if HAVE_ASYNC_LDS
  __builtin_amdgcn_global_load_async_to_lds_b128(
      (GLOBAL_AS v4i*)(v4i*)g, (LDS_AS v4i*)(v4i*)l, 0, 0);
#else
  *(Chunk16*)l = *(const Chunk16*)g;
#endif
}
__device__ __forceinline__ void stage_wait() {
#if HAVE_ASYNC_LDS
#  if __has_builtin(__builtin_amdgcn_s_wait_asynccnt)
  __builtin_amdgcn_s_wait_asynccnt(0);
#  else
  asm volatile("s_wait_asynccnt 0" ::: "memory");
#  endif
#endif
}

// ---- bf16 <-> f32 (round-to-nearest-even) ----
__device__ __forceinline__ u16 f2b(float f) {
  uint32_t u = __float_as_uint(f);
  uint32_t r = u + 0x7FFFu + ((u >> 16) & 1u);
  return (u16)(r >> 16);
}
__device__ __forceinline__ float b2f(u16 h) {
  return __uint_as_float(((uint32_t)h) << 16);
}
__device__ __forceinline__ float sigm(float x) { return 1.0f / (1.0f + __expf(-x)); }

// ---- WMMA tile loaders per CDNA5 ISA 7.12.2 (16-bit A 16x32 / B 32x16) ----
__device__ __forceinline__ v16bf load_a(const u16* rowptr_kk, int lane) {
  ABTile t;
  int o = (lane >> 4) << 3;                     // lanes 0-15: K 0..7,16..23 ; 16-31: K 8..15,24..31
  t.c[0] = *(const Chunk16*)(rowptr_kk + o);
  t.c[1] = *(const Chunk16*)(rowptr_kk + 16 + o);
  return t.v;
}
__device__ __forceinline__ v16bf load_b(const u16* colptr_kk, int lane) {
  ABTile t;
  int o = (lane >> 4) << 3;                     // lanes 0-15: K 0..7,16..23 ; 16-31: K 8..15,24..31
  t.c[0] = *(const Chunk16*)(colptr_kk + o);
  t.c[1] = *(const Chunk16*)(colptr_kk + 16 + o);
  return t.v;
}
#define WMMA_BF16(A, Bv, Cacc) \
  __builtin_amdgcn_wmma_f32_16x16x32_bf16(false, (A), false, (Bv), (short)0, (Cacc), false, false)

// ================= weight convert: f32 [N,K] -> bf16 [N,Kpad] (zero pad K) =================
__global__ void k_convert_w(const float* __restrict__ src, u16* __restrict__ dst,
                            int N, int K, int Kpad) {
  int idx = blockIdx.x * blockDim.x + threadIdx.x;
  if (idx >= N * Kpad) return;
  int n = idx / Kpad, kk = idx - n * Kpad;
  float v = (kk < K) ? src[(size_t)n * K + kk] : 0.0f;
  dst[idx] = f2b(v);
}

// ================= embedding gather: tokens -> bf16 [rows, EP_] =================
__global__ void k_gather_embed(const int* __restrict__ tok, const float* __restrict__ embed,
                               u16* __restrict__ X, int rows) {
  int idx = blockIdx.x * blockDim.x + threadIdx.x;
  if (idx >= rows * EP_) return;
  int r = idx / EP_, kk = idx - r * EP_;
  float v = 0.0f;
  if (kk < E_) v = embed[(size_t)tok[r] * E_ + kk];
  X[idx] = f2b(v);
}

// ================= encoder: h = sig(o)*tanh(sig(i)*tanh(g)), gates = X@W^T + b =================
// X bf16 [8192, EP_], W bf16 [2048, EP_]. All 8 waves of a block share one M-tile:
// stage the A panel (16 x EP_) in LDS once via async global->LDS, then WMMA from LDS.
__global__ void __launch_bounds__(256)
k_encoder(const u16* __restrict__ X, const u16* __restrict__ W, const float* __restrict__ bias,
          float* __restrict__ hf, u16* __restrict__ hb) {
  __shared__ u16 apanel[16 * EP_];                                // 10 KB
  const int tid  = threadIdx.x;
  const int lane = tid & 31;
  const int wave = (blockIdx.x * blockDim.x + tid) >> 5;          // 16384 waves
  const int mt = wave >> 5;                                       // 512 M-tiles (same for whole block)
  const int jt = wave & 31;                                       // 32  N-tiles (H)
  // ---- async stage A panel: 16*EP_ halfs = 640 chunks of 16B ----
  const int CPR = EP_ / 8;                                        // 40 chunks per row
  for (int c = tid; c < 16 * CPR; c += 256) {
    int row = c / CPR, col8 = (c - row * CPR) * 8;
    stage16(X + (size_t)(mt * 16 + row) * EP_ + col8, apanel + row * EP_ + col8);
  }
  stage_wait();
  __syncthreads();

  const int jcol = jt * 16 + (lane & 15);
  const u16* ap  = apanel + (lane & 15) * EP_;
  const u16* bpi = W + (size_t)(jcol)        * EP_;
  const u16* bpg = W + (size_t)(jcol + 1024) * EP_;
  const u16* bpo = W + (size_t)(jcol + 1536) * EP_;
  v8f ai = {0,0,0,0,0,0,0,0}, ag = ai, ao = ai;
  for (int kk = 0; kk < EP_; kk += 32) {
    __builtin_prefetch(bpi + kk + 64, 0, 1);
    v16bf a = load_a(ap + kk, lane);
    ai = WMMA_BF16(a, load_b(bpi + kk, lane), ai);
    ag = WMMA_BF16(a, load_b(bpg + kk, lane), ag);
    ao = WMMA_BF16(a, load_b(bpo + kk, lane), ao);
  }
  float bbi = bias[jcol], bbg = bias[jcol + 1024], bbo = bias[jcol + 1536];
  for (int r = 0; r < 8; ++r) {
    int row = mt * 16 + ((lane >> 4) << 3) + r;
    float c = sigm(ai[r] + bbi) * tanhf(ag[r] + bbg);
    float h = sigm(ao[r] + bbo) * tanhf(c);
    size_t off = (size_t)row * H_ + jcol;
    hf[off] = h; hb[off] = f2b(h);
  }
}

// ================= plain GEMM: out = X @ W^T ; X bf16 [8192,H_], W bf16 [H_,H_], out f32 =================
__global__ void __launch_bounds__(256)
k_proj(const u16* __restrict__ X, const u16* __restrict__ W, float* __restrict__ out) {
  __shared__ u16 apanel[16 * H_];                                 // 16 KB
  const int tid  = threadIdx.x;
  const int lane = tid & 31;
  const int wave = (blockIdx.x * blockDim.x + tid) >> 5;
  const int mt = wave >> 5, jt = wave & 31;
  const int CPR = H_ / 8;                                         // 64 chunks per row
  for (int c = tid; c < 16 * CPR; c += 256) {
    int row = c / CPR, col8 = (c - row * CPR) * 8;
    stage16(X + (size_t)(mt * 16 + row) * H_ + col8, apanel + row * H_ + col8);
  }
  stage_wait();
  __syncthreads();

  const u16* ap = apanel + (lane & 15) * H_;
  const u16* bp = W + (size_t)(jt * 16 + (lane & 15)) * H_;
  v8f acc = {0,0,0,0,0,0,0,0};
  for (int kk = 0; kk < H_; kk += 32) {
    __builtin_prefetch(bp + kk + 64, 0, 1);
    acc = WMMA_BF16(load_a(ap + kk, lane), load_b(bp + kk, lane), acc);
  }
  int jcol = jt * 16 + (lane & 15);
  for (int r = 0; r < 8; ++r)
    out[(size_t)(mt * 16 + ((lane >> 4) << 3) + r) * H_ + jcol] = acc[r];
}

// ================= the 128-step match-LSTM scan; 1 block == 16 independent batch rows =================
__global__ void __launch_bounds__(512)
k_scan(const float* __restrict__ sproj, const float* __restrict__ tproj,
       const float* __restrict__ hs,    const float* __restrict__ ht,
       const u16* __restrict__ wm,      const u16* __restrict__ wihm,
       const float* __restrict__ bm,    const float* __restrict__ w_e,
       const int* __restrict__ plen_g,  const int* __restrict__ hlen_g,
       float* __restrict__ hm_out) {
  extern __shared__ char smem[];
  u16*   hmb  = (u16*)smem;                                     // [16][512]  bf16 state   16KB
  u16*   mk   = (u16*)(smem + 16 * 512 * 2);                    // [16][1024] bf16 m_k     32KB
  float* mpro = (float*)(smem + 16 * 512 * 2 + 16 * 1024 * 2);  // [16][512]  f32          32KB
  float* alp  = (float*)((char*)mpro + 16 * 512 * 4);           // [16][128]  f32           8KB
  __shared__ float wel[H_];
  __shared__ int plen[16], hlen[16];

  const int tid = threadIdx.x, lane = tid & 31, wv = tid >> 5;  // 16 waves
  const int rb0 = blockIdx.x * 16;
  const int lcol = lane & 15;

  for (int i = tid; i < 16 * 512; i += 512) hmb[i] = 0;
  for (int i = tid; i < H_; i += 512) wel[i] = w_e[i];
  if (tid < 16) { plen[tid] = plen_g[rb0 + tid]; hlen[tid] = hlen_g[rb0 + tid]; }
  __syncthreads();

  for (int k = 0; k < ST_; ++k) {
    // ---- phase 1: mpro = h_m @ Wm^T + t_proj[:,k,:]   (WMMA, M=16) ----
    for (int rep = 0; rep < 2; ++rep) {
      int jt = wv + (rep << 4);
      const u16* ap = hmb + lcol * H_;
      const u16* bp = wm + (size_t)(jt * 16 + lcol) * H_;
      v8f acc = {0,0,0,0,0,0,0,0};
      for (int kk = 0; kk < H_; kk += 32) {
        __builtin_prefetch(bp + kk + 64, 0, 1);
        acc = WMMA_BF16(load_a(ap + kk, lane), load_b(bp + kk, lane), acc);
      }
      int j = jt * 16 + lcol;
      for (int r = 0; r < 8; ++r) {
        int bb = ((lane >> 4) << 3) + r;
        mpro[bb * H_ + j] = acc[r] + tproj[((size_t)(rb0 + bb) * ST_ + k) * H_ + j];
      }
    }
    __syncthreads();
    // ---- phase 2: e[b][s] = sum_h tanh(s_proj + mpro) * w_e  (wave per (b,s)) ----
    for (int i = 0; i < 128; ++i) {
      int idx = wv + (i << 4);
      int bb = idx >> 7, s = idx & 127;
      float e = -1e9f;
      if (s < plen[bb]) {
        const float* sp = sproj + ((size_t)(rb0 + bb) * SP_ + s) * H_;
        const float* mp = mpro + bb * H_;
        float part = 0.0f;
        for (int h = lane; h < H_; h += 32) part += tanhf(sp[h] + mp[h]) * wel[h];
        for (int off = 16; off; off >>= 1) part += __shfl_xor(part, off, 32);
        e = part;
      }
      if (lane == 0) alp[bb * SP_ + s] = e;
    }
    __syncthreads();
    // ---- phase 3: softmax over SP (wave per row) ----
    {
      int bb = wv;
      float v0 = alp[bb * SP_ + lane],      v1 = alp[bb * SP_ + lane + 32];
      float v2 = alp[bb * SP_ + lane + 64], v3 = alp[bb * SP_ + lane + 96];
      float m = fmaxf(fmaxf(v0, v1), fmaxf(v2, v3));
      for (int off = 16; off; off >>= 1) m = fmaxf(m, __shfl_xor(m, off, 32));
      v0 = __expf(v0 - m); v1 = __expf(v1 - m); v2 = __expf(v2 - m); v3 = __expf(v3 - m);
      float s4 = v0 + v1 + v2 + v3;
      for (int off = 16; off; off >>= 1) s4 += __shfl_xor(s4, off, 32);
      float inv = 1.0f / s4;
      alp[bb * SP_ + lane]      = v0 * inv; alp[bb * SP_ + lane + 32] = v1 * inv;
      alp[bb * SP_ + lane + 64] = v2 * inv; alp[bb * SP_ + lane + 96] = v3 * inv;
    }
    __syncthreads();
    // ---- phase 4: a_k = alpha @ h_s ; m_k = [a_k, h_t[:,k,:]] in bf16 ----
    {
      int h = tid;                                  // 512 threads == H_
      for (int bb = 0; bb < 16; ++bb) {
        const float* row = hs + (size_t)(rb0 + bb) * SP_ * H_ + h;
        const float* al  = alp + bb * SP_;
        float acc = 0.0f;
        int L = plen[bb];
        for (int s = 0; s < L; ++s) acc += al[s] * row[(size_t)s * H_];
        mk[bb * H2_ + h]      = f2b(acc);
        mk[bb * H2_ + H_ + h] = f2b(ht[((size_t)(rb0 + bb) * ST_ + k) * H_ + h]);
      }
    }
    __syncthreads();
    // ---- phase 5: match cell: gates = m_k @ W_ih_m^T (i,g,o only), update frozen-masked h_m ----
    for (int rep = 0; rep < 2; ++rep) {
      int jt = wv + (rep << 4);
      int j  = jt * 16 + lcol;
      const u16* ap  = mk + lcol * H2_;
      const u16* bpi = wihm + (size_t)(j)        * H2_;
      const u16* bpg = wihm + (size_t)(j + 1024) * H2_;
      const u16* bpo = wihm + (size_t)(j + 1536) * H2_;
      v8f ai = {0,0,0,0,0,0,0,0}, ag = ai, ao = ai;
      for (int kk = 0; kk < H2_; kk += 32) {
        __builtin_prefetch(bpi + kk + 64, 0, 1);
        v16bf a = load_a(ap + kk, lane);
        ai = WMMA_BF16(a, load_b(bpi + kk, lane), ai);
        ag = WMMA_BF16(a, load_b(bpg + kk, lane), ag);
        ao = WMMA_BF16(a, load_b(bpo + kk, lane), ao);
      }
      float bbi = bm[j], bbg = bm[j + 1024], bbo = bm[j + 1536];
      for (int r = 0; r < 8; ++r) {
        int bb = ((lane >> 4) << 3) + r;
        if (k < hlen[bb]) {
          float c = sigm(ai[r] + bbi) * tanhf(ag[r] + bbg);
          hmb[bb * H_ + j] = f2b(sigm(ao[r] + bbo) * tanhf(c));
        }
      }
    }
    __syncthreads();
  }
  for (int i = tid; i < 16 * 512; i += 512)
    hm_out[(size_t)(rb0 + (i >> 9)) * H_ + (i & 511)] = b2f(hmb[i]);
}

// ================= classifier head + log_softmax =================
__global__ void k_head(const float* __restrict__ hm, const float* __restrict__ fcW,
                       const float* __restrict__ fcb, float* __restrict__ out) {
  int b = blockIdx.x * blockDim.x + threadIdx.x;
  if (b >= B_) return;
  float l[C_];
  for (int c = 0; c < C_; ++c) {
    float acc = fcb[c];
    const float* w = fcW + c * H_;
    const float* h = hm + (size_t)b * H_;
    for (int i = 0; i < H_; ++i) acc += h[i] * w[i];
    l[c] = acc;
  }
  float m = fmaxf(l[0], fmaxf(l[1], l[2]));
  float lse = m + __logf(__expf(l[0] - m) + __expf(l[1] - m) + __expf(l[2] - m));
  for (int c = 0; c < C_; ++c) out[b * C_ + c] = l[c] - lse;
}

// ================= launcher =================
extern "C" void kernel_launch(void* const* d_in, const int* in_sizes, int n_in,
                              void* d_out, int out_size, void* d_ws, size_t ws_size,
                              hipStream_t stream) {
  (void)in_sizes; (void)n_in; (void)out_size; (void)ws_size;
  const int*   premise = (const int*)  d_in[0];
  const int*   plen    = (const int*)  d_in[1];
  const int*   hypoth  = (const int*)  d_in[2];
  const int*   hlen    = (const int*)  d_in[3];
  const float* embed   = (const float*)d_in[4];
  const float* w_e     = (const float*)d_in[5];
  const float* Ws      = (const float*)d_in[6];
  const float* Wt      = (const float*)d_in[7];
  const float* Wm      = (const float*)d_in[8];
  const float* W_ih_p  = (const float*)d_in[9];
  const float* b_p     = (const float*)d_in[10];
  const float* W_ih_h  = (const float*)d_in[11];
  const float* b_h     = (const float*)d_in[12];
  const float* W_ih_m  = (const float*)d_in[13];
  const float* b_m     = (const float*)d_in[14];
  const float* fc_W    = (const float*)d_in[15];
  const float* fc_b    = (const float*)d_in[16];
  float* out = (float*)d_out;

  char* ws = (char*)d_ws;
  size_t o = 0;
  auto take = [&](size_t bytes) -> char* {
    char* p = ws + o; o += (bytes + 255) & ~(size_t)255; return p;
  };
  u16*   wihp_b = (u16*)  take((size_t)G4_ * EP_ * 2);
  u16*   wihh_b = (u16*)  take((size_t)G4_ * EP_ * 2);
  u16*   ws_b   = (u16*)  take((size_t)H_ * H_ * 2);
  u16*   wt_b   = (u16*)  take((size_t)H_ * H_ * 2);
  u16*   wm_b   = (u16*)  take((size_t)H_ * H_ * 2);
  u16*   wihm_b = (u16*)  take((size_t)G4_ * H2_ * 2);
  u16*   Xp     = (u16*)  take((size_t)B_ * SP_ * EP_ * 2);
  u16*   Xh     = (u16*)  take((size_t)B_ * ST_ * EP_ * 2);
  float* hs_f   = (float*)take((size_t)B_ * SP_ * H_ * 4);
  u16*   hs_b   = (u16*)  take((size_t)B_ * SP_ * H_ * 2);
  float* ht_f   = (float*)take((size_t)B_ * ST_ * H_ * 4);
  u16*   ht_b   = (u16*)  take((size_t)B_ * ST_ * H_ * 2);
  float* sproj  = (float*)take((size_t)B_ * SP_ * H_ * 4);
  float* tproj  = (float*)take((size_t)B_ * ST_ * H_ * 4);
  float* hm     = (float*)take((size_t)B_ * H_ * 4);

  // weights -> bf16 (natural [N,K] layout == WMMA B layout; only K-pad for E)
  { int t = G4_ * EP_; k_convert_w<<<(t + 255) / 256, 256, 0, stream>>>(W_ih_p, wihp_b, G4_, E_, EP_); }
  { int t = G4_ * EP_; k_convert_w<<<(t + 255) / 256, 256, 0, stream>>>(W_ih_h, wihh_b, G4_, E_, EP_); }
  { int t = H_ * H_;   k_convert_w<<<(t + 255) / 256, 256, 0, stream>>>(Ws, ws_b, H_, H_, H_); }
  { int t = H_ * H_;   k_convert_w<<<(t + 255) / 256, 256, 0, stream>>>(Wt, wt_b, H_, H_, H_); }
  { int t = H_ * H_;   k_convert_w<<<(t + 255) / 256, 256, 0, stream>>>(Wm, wm_b, H_, H_, H_); }
  { int t = G4_ * H2_; k_convert_w<<<(t + 255) / 256, 256, 0, stream>>>(W_ih_m, wihm_b, G4_, H2_, H2_); }
  // embedding gathers -> bf16 activation matrices
  { int t = B_ * SP_ * EP_; k_gather_embed<<<(t + 255) / 256, 256, 0, stream>>>(premise, embed, Xp, B_ * SP_); }
  { int t = B_ * ST_ * EP_; k_gather_embed<<<(t + 255) / 256, 256, 0, stream>>>(hypoth, embed, Xh, B_ * ST_); }
  // encoders (fused LSTM-gate epilogue): 512 Mtiles * 32 Ntiles waves / 8 waves per block
  k_encoder<<<2048, 256, 0, stream>>>(Xp, wihp_b, b_p, hs_f, hs_b);
  k_encoder<<<2048, 256, 0, stream>>>(Xh, wihh_b, b_h, ht_f, ht_b);
  // projections
  k_proj<<<2048, 256, 0, stream>>>(hs_b, ws_b, sproj);
  k_proj<<<2048, 256, 0, stream>>>(ht_b, wt_b, tproj);
  // scan: 4 blocks x 16 batch rows, 88KB dynamic LDS (320KB/WGP on gfx1250)
  size_t smem = (size_t)16 * 512 * 2 + 16 * 1024 * 2 + 16 * 512 * 4 + 16 * 128 * 4;
  k_scan<<<4, 512, smem, stream>>>(sproj, tproj, hs_f, ht_f, wm_b, wihm_b, b_m, w_e, plen, hlen, hm);
  // head
  k_head<<<1, 64, 0, stream>>>(hm, fc_W, fc_b, out);
}
